// Model_30760555774480
// MI455X (gfx1250) — compile-verified, matrix operand order checked by reading, not run on hardware
//
#include <hip/hip_runtime.h>
#include <hip/hip_bf16.h>

typedef __bf16 bf16_t;
typedef __attribute__((ext_vector_type(16))) __bf16 v16bf;
typedef __attribute__((ext_vector_type(8)))  float  v8f;

struct __attribute__((aligned(16))) U128 { unsigned int u[4]; };
struct __attribute__((aligned(16))) F128 { float f[4]; };
struct __attribute__((aligned(8)))  U64  { unsigned int u[2]; };
union  FragAB { v16bf v; U128 q[2]; };

#define NCLS 10
#define DIMD 3072
#define DIMH 512
#define DIMZ 256
#define BATCH 2048

#define BM  128
#define BN  64
#define BKK 64          // K-tile (two WMMA K=32 steps per staged tile)
#define LDT 72          // padded LDS stride in bf16 (144 B, 16B aligned)
#define ACH 8           // A chunks (float4/bf16x4) per thread: 128*64/(4*256)
#define BCH 4           // B chunks (float4) per thread:        64*64/(4*256)

// ---------------- routing kernels ----------------
__global__ void k_zero(int* counts) {
    if (threadIdx.x < NCLS) counts[threadIdx.x] = 0;
}
__global__ void k_hist(const int* __restrict__ label, int* counts) {
    int i = blockIdx.x * blockDim.x + threadIdx.x;
    if (i < BATCH) atomicAdd(&counts[label[i]], 1);
}
__global__ void k_scan(const int* __restrict__ counts, int* offs, int* cursor) {
    if (threadIdx.x == 0 && blockIdx.x == 0) {
        int s = 0;
        for (int c = 0; c < NCLS; ++c) { offs[c] = s; cursor[c] = s; s += counts[c]; }
    }
}
__global__ void k_fill(const int* __restrict__ label, int* cursor, int* rowidx) {
    int i = blockIdx.x * blockDim.x + threadIdx.x;
    if (i < BATCH) { int p = atomicAdd(&cursor[label[i]], 1); rowidx[p] = i; }
}

// ---------------- routed, pipelined WMMA GEMM ----------------
// Rows past cnt are clamped to cnt-1 (duplicate of a valid row, finite data).
// They only affect D rows >= cnt, which the epilogue never stores, so no
// zero-masking is needed anywhere in the hot loop.
template<bool AGATHER, bool RELU, bool SCATTER>
__global__ __launch_bounds__(256) void moe_gemm(
    const float*  __restrict__ Af32, const bf16_t* __restrict__ A16,
    const float*  __restrict__ W,    const float*  __restrict__ bias,
    const int*    __restrict__ rowidx, const int* __restrict__ offs,
    const int*    __restrict__ counts,
    bf16_t*       __restrict__ Out16, float* __restrict__ OutF32,
    int K, int N)
{
    const int c   = blockIdx.z;
    const int cnt = counts[c];
    const int m0  = blockIdx.y * BM;
    if (m0 >= cnt) return;                 // block-uniform early exit
    const int n0  = blockIdx.x * BN;
    const int off = offs[c];

    const int t    = threadIdx.x;
    const int lane = t & 31;
    const int wave = t >> 5;               // 8 waves
    const int wm   = wave >> 1;            // 0..3 -> 32-row strip
    const int wn   = wave & 1;             // 0..1 -> 32-col strip

    __shared__ __attribute__((aligned(16))) bf16_t lA[2][BM * LDT];
    __shared__ __attribute__((aligned(16))) bf16_t lB[2][BN * LDT];

    const float* Wc = W    + (size_t)c * K * N + n0;
    const float* bc = bias + (size_t)c * N + n0;

    // ---- precompute per-thread staging addresses (hoisted out of K loop) ----
    size_t abase[ACH];     // source element offset of (row, kk); k0 added later
    int    aoff [ACH];     // LDS offset
    #pragma unroll
    for (int i = 0; i < ACH; ++i) {
        int cidx = t + i * 256;            // 0..2047
        int m  = cidx >> 4;                // 16 float4-chunks per 64-wide row
        int kk = (cidx & 15) * 4;
        int gm  = m0 + m;
        int gmc = (gm < cnt) ? gm : (cnt - 1);   // clamp: always valid memory
        if (AGATHER) { int gr = rowidx[off + gmc]; abase[i] = (size_t)gr * K + kk; }
        else         {                     abase[i] = (size_t)(off + gmc) * K + kk; }
        aoff[i] = m * LDT + kk;
    }
    int bkk[BCH], bnn[BCH];
    #pragma unroll
    for (int j = 0; j < BCH; ++j) {
        int cidx = t + j * 256;            // 0..1023
        bkk[j] = cidx >> 4;                // k within tile
        bnn[j] = (cidx & 15) * 4;          // n within tile
    }

    // ---- register prefetch buffers ----
    F128 raf[ACH];     // fp32 A path
    U64  rah[ACH];     // bf16 A path
    F128 rb [BCH];

    auto load_tile = [&](int k0) {
        #pragma unroll
        for (int i = 0; i < ACH; ++i) {
            if (AGATHER) raf[i] = *(const F128*)(Af32 + abase[i] + k0);
            else         rah[i] = *(const U64*) (A16  + abase[i] + k0);
        }
        #pragma unroll
        for (int j = 0; j < BCH; ++j)
            rb[j] = *(const F128*)(Wc + (size_t)(k0 + bkk[j]) * N + bnn[j]);
    };

    auto store_tile = [&](int buf) {
        #pragma unroll
        for (int i = 0; i < ACH; ++i) {
            if (AGATHER) {
                union { bf16_t h[4]; U64 u; } pk;
                pk.h[0] = (bf16_t)raf[i].f[0]; pk.h[1] = (bf16_t)raf[i].f[1];
                pk.h[2] = (bf16_t)raf[i].f[2]; pk.h[3] = (bf16_t)raf[i].f[3];
                *(U64*)&lA[buf][aoff[i]] = pk.u;
            } else {
                *(U64*)&lA[buf][aoff[i]] = rah[i];
            }
        }
        #pragma unroll
        for (int j = 0; j < BCH; ++j) {
            F128 v = rb[j];
            lB[buf][(bnn[j] + 0) * LDT + bkk[j]] = (bf16_t)v.f[0];
            lB[buf][(bnn[j] + 1) * LDT + bkk[j]] = (bf16_t)v.f[1];
            lB[buf][(bnn[j] + 2) * LDT + bkk[j]] = (bf16_t)v.f[2];
            lB[buf][(bnn[j] + 3) * LDT + bkk[j]] = (bf16_t)v.f[3];
        }
    };

    v8f acc[2][2] = {};
    const int rsel = (lane >= 16) ? 8 : 0; // K half-split per half-wave
    const int lrow = lane & 15;

    auto compute = [&](int buf) {
        #pragma unroll
        for (int h = 0; h < 2; ++h) {      // two K=32 sub-steps
            FragAB a[2], b[2];
            #pragma unroll
            for (int i = 0; i < 2; ++i) {
                const bf16_t* p = &lA[buf][(wm * 32 + i * 16 + lrow) * LDT + h * 32 + rsel];
                a[i].q[0] = *(const U128*)p;
                a[i].q[1] = *(const U128*)(p + 16);
            }
            #pragma unroll
            for (int j = 0; j < 2; ++j) {
                const bf16_t* p = &lB[buf][(wn * 32 + j * 16 + lrow) * LDT + h * 32 + rsel];
                b[j].q[0] = *(const U128*)p;
                b[j].q[1] = *(const U128*)(p + 16);
            }
            #pragma unroll
            for (int i = 0; i < 2; ++i)
                #pragma unroll
                for (int j = 0; j < 2; ++j)
                    acc[i][j] = __builtin_amdgcn_wmma_f32_16x16x32_bf16(
                        false, a[i].v, false, b[j].v, (short)0, acc[i][j], false, false);
        }
    };

    // ---- software pipeline: regs(s+1) overlap WMMA(s); 1 barrier/step ----
    const int NS = K / BKK;
    load_tile(0);
    for (int s = 0; s < NS; ++s) {
        store_tile(s & 1);
        __syncthreads();
        if (s + 1 < NS) load_tile((s + 1) * BKK);
        compute(s & 1);
    }

    // ---- epilogue: bias (+ReLU), compact bf16 or fp32 scatter ----
    const int mhi = (lane >= 16) ? 8 : 0;
    float bv[2] = { bc[wn * 32 + lrow], bc[wn * 32 + 16 + lrow] };
    #pragma unroll
    for (int i = 0; i < 2; ++i)
        #pragma unroll
        for (int r = 0; r < 8; ++r) {
            int row = wm * 32 + i * 16 + mhi + r;
            int gm  = m0 + row;
            if (gm < cnt) {
                size_t obase = SCATTER ? (size_t)rowidx[off + gm] * N
                                       : (size_t)(off + gm) * N;
                #pragma unroll
                for (int j = 0; j < 2; ++j) {
                    int col = wn * 32 + j * 16 + lrow;
                    float v = acc[i][j][r] + bv[j];
                    if (RELU) v = fmaxf(v, 0.0f);
                    if (SCATTER) OutF32[obase + n0 + col] = v;
                    else         Out16 [obase + n0 + col] = (bf16_t)v;
                }
            }
        }
}

// ---------------- deterministic MSE reduction ----------------
__global__ void loss_partial(const float* __restrict__ outImg,
                             const float* __restrict__ img,
                             float* __restrict__ partials, long long n) {
    __shared__ float s[256];
    float acc = 0.0f;
    for (long long i = (long long)blockIdx.x * 256 + threadIdx.x; i < n;
         i += (long long)gridDim.x * 256) {
        float d = outImg[i] - img[i];
        acc += d * d;
    }
    s[threadIdx.x] = acc; __syncthreads();
    for (int st = 128; st > 0; st >>= 1) {
        if (threadIdx.x < st) s[threadIdx.x] += s[threadIdx.x + st];
        __syncthreads();
    }
    if (threadIdx.x == 0) partials[blockIdx.x] = s[0];
}
__global__ void loss_final(const float* __restrict__ partials, int P,
                           float* __restrict__ out, float inv) {
    __shared__ float s[256];
    float acc = 0.0f;
    for (int i = threadIdx.x; i < P; i += 256) acc += partials[i];
    s[threadIdx.x] = acc; __syncthreads();
    for (int st = 128; st > 0; st >>= 1) {
        if (threadIdx.x < st) s[threadIdx.x] += s[threadIdx.x + st];
        __syncthreads();
    }
    if (threadIdx.x == 0) out[0] = s[0] * inv;
}

extern "C" void kernel_launch(void* const* d_in, const int* in_sizes, int n_in,
                              void* d_out, int out_size, void* d_ws, size_t ws_size,
                              hipStream_t stream) {
    const float* img    = (const float*)d_in[0];
    const int*   label  = (const int*)  d_in[1];
    const float* enc_w1 = (const float*)d_in[2];
    const float* enc_b1 = (const float*)d_in[3];
    const float* enc_w2 = (const float*)d_in[4];
    const float* enc_b2 = (const float*)d_in[5];
    const float* dec_w1 = (const float*)d_in[6];
    const float* dec_b1 = (const float*)d_in[7];
    const float* dec_w2 = (const float*)d_in[8];
    const float* dec_b2 = (const float*)d_in[9];

    float* loss   = (float*)d_out;   // d_out[0] = loss
    float* outImg = loss + 1;        // d_out[1..] = out_img [B,D]

    char* w = (char*)d_ws;
    auto take = [&](size_t bytes, size_t align) -> void* {
        size_t a = ((size_t)w + align - 1) & ~(align - 1);
        w = (char*)(a + bytes);
        return (void*)a;
    };
    int*    counts   = (int*)take(NCLS * 4, 256);
    int*    offs     = (int*)take(NCLS * 4, 256);
    int*    cursor   = (int*)take(NCLS * 4, 256);
    int*    rowidx   = (int*)take(BATCH * 4, 256);
    bf16_t* h16      = (bf16_t*)take((size_t)BATCH * DIMH * 2, 256);
    bf16_t* z16      = (bf16_t*)take((size_t)BATCH * DIMZ * 2, 256);
    bf16_t* g16      = (bf16_t*)take((size_t)BATCH * DIMH * 2, 256);
    float*  partials = (float*)take(1024 * 4, 256);

    // 1) route rows by label
    k_zero<<<1, 32, 0, stream>>>(counts);
    k_hist<<<(BATCH + 255) / 256, 256, 0, stream>>>(label, counts);
    k_scan<<<1, 1, 0, stream>>>(counts, offs, cursor);
    k_fill<<<(BATCH + 255) / 256, 256, 0, stream>>>(label, cursor, rowidx);

    dim3 blk(256);
    // 2) four routed GEMMs (WMMA bf16), ~10x fewer FLOPs than the reference
    moe_gemm<true,  true,  false><<<dim3(DIMH / BN, BATCH / BM, NCLS), blk, 0, stream>>>(
        img, nullptr, enc_w1, enc_b1, rowidx, offs, counts, h16, nullptr, DIMD, DIMH);
    moe_gemm<false, false, false><<<dim3(DIMZ / BN, BATCH / BM, NCLS), blk, 0, stream>>>(
        nullptr, h16, enc_w2, enc_b2, rowidx, offs, counts, z16, nullptr, DIMH, DIMZ);
    moe_gemm<false, true,  false><<<dim3(DIMH / BN, BATCH / BM, NCLS), blk, 0, stream>>>(
        nullptr, z16, dec_w1, dec_b1, rowidx, offs, counts, g16, nullptr, DIMZ, DIMH);
    moe_gemm<false, false, true ><<<dim3(DIMD / BN, BATCH / BM, NCLS), blk, 0, stream>>>(
        nullptr, g16, dec_w2, dec_b2, rowidx, offs, counts, nullptr, outImg, DIMH, DIMD);

    // 3) loss = mean((out - img)^2), deterministic two-stage reduction
    long long ntot = (long long)BATCH * DIMD;
    loss_partial<<<1024, 256, 0, stream>>>(outImg, img, partials, ntot);
    loss_final<<<1, 256, 0, stream>>>(partials, 1024, loss, 1.0f / (float)ntot);
}
